// MultiHeadAttention_13280038879526
// MI455X (gfx1250) — matrix operand order repaired and checked by reading, not compiled, not run on hardware
//
#include <hip/hip_runtime.h>

// ---------------------------------------------------------------------------
// MHA forward, MI455X (gfx1250).  bf16 WMMA + double-buffered TDM pipelines.
// B=2, S=2048, D=1024, H=16, HD=64.
// ---------------------------------------------------------------------------

typedef __attribute__((ext_vector_type(16))) __bf16 v16bf;
typedef __attribute__((ext_vector_type(8)))  float  v8f;
typedef __attribute__((ext_vector_type(2)))  float  floatx2;
typedef __attribute__((ext_vector_type(2)))  __bf16 bf16x2;

union Frag {                 // one 16x32 bf16 WMMA operand (8 VGPRs / lane)
    unsigned u[8];
    v16bf    v;
};

#define BB 2
#define SS 2048
#define DD 1024
#define HH 16
#define HD 64

// native packed f32 -> bf16 conversion (v_cvt_pk_bf16_f32)
__device__ __forceinline__ unsigned packcvt(float lo, float hi) {
    floatx2 f = {lo, hi};
    bf16x2 b = __builtin_convertvector(f, bf16x2);
    union { bf16x2 b; unsigned u; } u; u.b = b; return u.u;
}
__device__ __forceinline__ unsigned short f2bf(float f) {
    union { __bf16 b; unsigned short s; } u; u.b = (__bf16)f; return u.s;
}

// A-matrix 16-bit 16x32 layout (contiguous-K per lane-half): VGPR v holds K pair starting at:
__device__ __forceinline__ int kbaseA(int v, int half) {
    return 2 * v + (half ? 16 : 0);
}
// B-matrix 16-bit 32x16 layout (contiguous-K per lane-half):
__device__ __forceinline__ int kbaseB(int v, int half) {
    return 2 * v + (half ? 16 : 0);
}

__device__ __forceinline__ v8f wmma_bf16(const Frag& a, const Frag& b, v8f c) {
    return __builtin_amdgcn_wmma_f32_16x16x32_bf16(false, a.v, false, b.v,
                                                   (short)0, c, false, false);
}

// ---------------------------------------------------------------------------
// Tensor Data Mover: 2-D tile load Global -> LDS (ISA 08_async_tensor §8).
// ---------------------------------------------------------------------------
#if __has_builtin(__builtin_amdgcn_tensor_load_to_lds)
#define HAVE_TDM 1
#else
#define HAVE_TDM 0
#endif

#if HAVE_TDM
typedef unsigned int u32x4 __attribute__((ext_vector_type(4)));
typedef int          i32x4 __attribute__((ext_vector_type(4)));
typedef int          i32x8 __attribute__((ext_vector_type(8)));

__device__ __forceinline__ unsigned lds_off(const void* p) {
    return (unsigned)(size_t)p;      // low 32 bits of flat LDS addr = LDS offset
}

__device__ __forceinline__ void tdm_load_2d(unsigned lds_byte_off, const void* gptr,
                                            unsigned dsz_code,          // 0/1/2/3 = 1/2/4/8B
                                            unsigned t_d0, unsigned t_d1,
                                            unsigned stride0,           // elements
                                            unsigned tile_d0, unsigned tile_d1) {
    unsigned long long ga = (unsigned long long)(size_t)gptr;
    u32x4 g0;
    g0[0] = 1u;                                                // count=1, user mode
    g0[1] = lds_byte_off;                                      // lds_addr
    g0[2] = (unsigned)ga;                                      // global_addr[31:0]
    g0[3] = (unsigned)((ga >> 32) & 0x01FFFFFFu) | (2u << 30); // addr[56:32], type=2
    i32x8 g1;
    g1[0] = (int)(dsz_code << 16);                             // mask=0, data_size
    g1[1] = (int)((t_d0 & 0xFFFFu) << 16);                     // tensor_dim0 lo
    g1[2] = (int)(((t_d0 >> 16) & 0xFFFFu) | ((t_d1 & 0xFFFFu) << 16));
    g1[3] = (int)(((t_d1 >> 16) & 0xFFFFu) | (tile_d0 << 16));
    g1[4] = (int)(tile_d1 & 0xFFFFu);                          // tile_dim1, tile_dim2=0
    g1[5] = (int)stride0;                                      // dim0_stride lo
    g1[6] = 0;
    g1[7] = 0;
    i32x4 gz = {0, 0, 0, 0};
#if __clang_major__ >= 23
    i32x8 gz8 = {0, 0, 0, 0, 0, 0, 0, 0};
    __builtin_amdgcn_tensor_load_to_lds(g0, g1, gz, gz, gz8, 0);
#else
    __builtin_amdgcn_tensor_load_to_lds(g0, g1, gz, gz, 0);
#endif
}
#endif

// ---------------------------------------------------------------------------
// Kernel 0: weight pre-pass.  fp32 W[D,D] -> k-pair-interleaved bf16x2 u32.
// ---------------------------------------------------------------------------
__global__ void __launch_bounds__(256)
mha_cvt_wpair(const float* __restrict__ W, unsigned* __restrict__ O) {
    int idx = blockIdx.x * 256 + threadIdx.x;       // 0 .. (D/2)*D-1
    int kp = idx >> 10, n = idx & (DD - 1);
    O[idx] = packcvt(W[(2 * kp) * DD + n], W[(2 * kp + 1) * DD + n]);
}

// ---------------------------------------------------------------------------
// Kernel 1: fused QKV projection, 64x64 block tile, double-buffered weight
// tiles via TDM.  z==0: Q (scaled, [B,H,S,HD]); z==1: K; z==2: V^T.
// grid = (B*S/64, D/64, 3), block = 128.
// ---------------------------------------------------------------------------
__global__ void __launch_bounds__(128)
mha_proj_qkv(const float* __restrict__ Xq, const float* __restrict__ Xk,
             const float* __restrict__ Xv,
             const unsigned* __restrict__ Wqp, const unsigned* __restrict__ Wkp,
             const unsigned* __restrict__ Wvp,
             unsigned short* __restrict__ Qo, unsigned short* __restrict__ Ko,
             unsigned short* __restrict__ Vo) {
    const int tid  = threadIdx.x;
    const int wid  = tid >> 5;
    const int lane = tid & 31;
    const int half = lane >> 4;
    const int lm   = lane & 15;

    const int m0 = blockIdx.x * 64 + wid * 16;
    const int nb = blockIdx.y * 64;
    const int z  = blockIdx.z;

    const float* X; const unsigned* Wp; unsigned short* O; float scale;
    if (z == 0)      { X = Xq; Wp = Wqp; O = Qo; scale = 0.125f; }  // HD^-0.5
    else if (z == 1) { X = Xk; Wp = Wkp; O = Ko; scale = 1.0f; }
    else             { X = Xv; Wp = Wvp; O = Vo; scale = 1.0f; }

    __shared__ unsigned wt[2][16 * 64];   // double-buffered 16 kp x 64 n tiles

    v8f acc[4];
#pragma unroll
    for (int f = 0; f < 4; ++f) acc[f] = (v8f){0,0,0,0,0,0,0,0};

#if HAVE_TDM
    // prologue: start DMA of tile 0 (each wave moves 4 kp-rows of its slice)
    tdm_load_2d(lds_off(&wt[0][0]) + wid * 4 * 64 * 4,
                Wp + (wid * 4) * DD + nb,
                2, DD, DD / 2, DD, 64, 4);
#endif

#pragma unroll 1
    for (int k0 = 0; k0 < DD; k0 += 32) {
        const int cur = (k0 >> 5) & 1;
        const unsigned* wcur = wt[cur];
#if HAVE_TDM
        if (k0 + 32 < DD) {
            tdm_load_2d(lds_off(&wt[cur ^ 1][0]) + wid * 4 * 64 * 4,
                        Wp + (((k0 + 32) >> 1) + wid * 4) * DD + nb,
                        2, DD, DD / 2, DD, 64, 4);
            __builtin_amdgcn_s_wait_tensorcnt(1);   // older tile landed
        } else {
            __builtin_amdgcn_s_wait_tensorcnt(0);
        }
#else
#pragma unroll
        for (int i = 0; i < 8; ++i) {
            int idx = i * 128 + tid;
            wt[cur][idx] = Wp[((k0 >> 1) + (idx >> 6)) * DD + nb + (idx & 63)];
        }
#endif
        __syncthreads();

        if (k0 + 32 < DD)
            __builtin_prefetch(X + (m0 + lm) * DD + k0 + 32, 0, 3);

        // ---- A fragment: 16 fp32 -> bf16 (packed cvt) ----
        Frag a;
#pragma unroll
        for (int v = 0; v < 8; ++v) {
            floatx2 f2 = *(const floatx2*)(X + (m0 + lm) * DD + k0 + kbaseA(v, half));
            a.u[v] = packcvt(f2.x, f2.y);
        }
        // ---- 4 B fragments from LDS, 4 WMMAs ----
#pragma unroll
        for (int f = 0; f < 4; ++f) {
            Frag b;
#pragma unroll
            for (int v = 0; v < 8; ++v)
                b.u[v] = wcur[(v + half * 8) * 64 + f * 16 + lm];
            acc[f] = wmma_bf16(a, b, acc[f]);
        }
        __syncthreads();    // all reads done before this buffer is re-filled
    }

    // ---- write out ----
#pragma unroll
    for (int f = 0; f < 4; ++f) {
        const int n  = nb + f * 16 + lm;
        const int h  = n >> 6;
        const int hd = n & 63;
#pragma unroll
        for (int r = 0; r < 8; ++r) {
            int row = r + (half ? 8 : 0);
            int rg  = m0 + row;
            int bb  = rg >> 11;
            int s   = rg & (SS - 1);
            unsigned short val = f2bf(acc[f][r] * scale);
            if (z == 2) O[((bb * HH + h) * HD + hd) * SS + s] = val;   // V^T
            else        O[((bb * HH + h) * SS + s) * HD + hd] = val;   // Q,K
        }
    }
}

// ---------------------------------------------------------------------------
// Kernel 2: flash attention.  4 waves/block share (b,h); K and V^T tiles are
// double-buffered through the TDM (2 descriptors in flight per buffer).
// grid = B*H*(S/16)/4 = 1024 blocks, block = 128.
// ---------------------------------------------------------------------------
__global__ void __launch_bounds__(128)
mha_attn(const unsigned short* __restrict__ Q, const unsigned short* __restrict__ K,
         const unsigned short* __restrict__ Vt, unsigned short* __restrict__ C) {
    const int tid  = threadIdx.x;
    const int wid  = tid >> 5;
    const int lane = tid & 31;
    const int half = lane >> 4;
    const int lm   = lane & 15;

    const int t  = blockIdx.x * 4 + wid;        // 4 consecutive qt share (b,h)
    const int qt = t & (SS / 16 - 1);
    const int h  = (t >> 7) & (HH - 1);
    const int b  = t >> 11;

    const long base = ((long)(b * HH + h)) * SS * HD;
    const unsigned short* Qp  = Q  + base;
    const unsigned short* Kp  = K  + base;
    const unsigned short* Vtp = Vt + base;      // [HD][S] layout

    __shared__ unsigned short kt[2][32 * 64];   // K tiles  [key][hd]
    __shared__ unsigned short vtt[2][64 * 32];  // V^T tiles [hd][key]
    __shared__ unsigned short pb[4][16 * 32];   // per-wave P tile
    unsigned short* pw = pb[wid];

    // ---- Q fragments (16x64 tile = two 16x32 A-frags), already bf16 ----
    Frag qa[2];
    const int qrow = qt * 16 + lm;
#pragma unroll
    for (int c = 0; c < 2; ++c)
#pragma unroll
        for (int v = 0; v < 8; ++v)
            qa[c].u[v] = *(const unsigned*)(Qp + (long)qrow * HD + c * 32 + kbaseA(v, half));

    float mrun[8], lrun[8];
#pragma unroll
    for (int r = 0; r < 8; ++r) { mrun[r] = -1.0e30f; lrun[r] = 0.0f; }
    v8f oc[4];
#pragma unroll
    for (int f = 0; f < 4; ++f) oc[f] = (v8f){0,0,0,0,0,0,0,0};

#if HAVE_TDM
    // prologue: DMA K/V^T tiles for the first 32 keys into buffer 0
    tdm_load_2d(lds_off(&kt[0][0]) + wid * 8 * 64 * 2,
                Kp + (wid * 8) * HD, 1, HD, SS, HD, 64, 8);
    tdm_load_2d(lds_off(&vtt[0][0]) + wid * 16 * 32 * 2,
                Vtp + (wid * 16) * SS, 1, SS, HD, SS, 32, 16);
#endif

#pragma unroll 1
    for (int kb2 = 0; kb2 < SS; kb2 += 32) {
        const int cur = (kb2 >> 5) & 1;
        const unsigned short* kc = kt[cur];
        const unsigned short* vc = vtt[cur];
#if HAVE_TDM
        if (kb2 + 32 < SS) {
            tdm_load_2d(lds_off(&kt[cur ^ 1][0]) + wid * 8 * 64 * 2,
                        Kp + (kb2 + 32 + wid * 8) * HD, 1, HD, SS, HD, 64, 8);
            tdm_load_2d(lds_off(&vtt[cur ^ 1][0]) + wid * 16 * 32 * 2,
                        Vtp + (wid * 16) * SS + kb2 + 32, 1, SS, HD, SS, 32, 16);
            __builtin_amdgcn_s_wait_tensorcnt(2);   // previous K/V pair landed
        } else {
            __builtin_amdgcn_s_wait_tensorcnt(0);
        }
#else
#pragma unroll
        for (int i = 0; i < 8; ++i) {   // K tile: 2048 ushorts contiguous
            int idx = i * 128 + tid;
            ((unsigned*)kc)[idx] = *(const unsigned*)(Kp + kb2 * HD + idx * 2);
        }
#pragma unroll
        for (int i = 0; i < 8; ++i) {   // V^T tile: 64 rows x 32
            int idx = i * 128 + tid;
            ((unsigned*)vc)[idx] =
                *(const unsigned*)(Vtp + (idx >> 4) * SS + kb2 + (idx & 15) * 2);
        }
#endif
        __syncthreads();

        // ---- logits S[16 x 32] = Q x K^T (two 16x16 frags) ----
        v8f s[2];
#pragma unroll
        for (int g = 0; g < 2; ++g) {
            Frag b0, b1;
            const int krow = (g * 16 + lm) * 64;     // B lane = key column
#pragma unroll
            for (int v = 0; v < 8; ++v) {
                int kk = kbaseB(v, half);            // contraction over hd
                b0.u[v] = *(const unsigned*)(kc + krow + kk);
                b1.u[v] = *(const unsigned*)(kc + krow + 32 + kk);
            }
            v8f zz = (v8f){0,0,0,0,0,0,0,0};
            zz   = wmma_bf16(qa[0], b0, zz);
            s[g] = wmma_bf16(qa[1], b1, zz);
        }

        // ---- online softmax (width-16 butterflies per lane-half) ----
#pragma unroll
        for (int r = 0; r < 8; ++r) {
            float mx = fmaxf(s[0][r], s[1][r]);
            mx = fmaxf(mx, __shfl_xor(mx, 8, 16));
            mx = fmaxf(mx, __shfl_xor(mx, 4, 16));
            mx = fmaxf(mx, __shfl_xor(mx, 2, 16));
            mx = fmaxf(mx, __shfl_xor(mx, 1, 16));
            float mn = fmaxf(mrun[r], mx);
            float sc = __expf(mrun[r] - mn);
            float p0 = __expf(s[0][r] - mn);
            float p1 = __expf(s[1][r] - mn);
            float rs = p0 + p1;
            rs += __shfl_xor(rs, 8, 16);
            rs += __shfl_xor(rs, 4, 16);
            rs += __shfl_xor(rs, 2, 16);
            rs += __shfl_xor(rs, 1, 16);
            lrun[r] = lrun[r] * sc + rs;
            mrun[r] = mn;
            oc[0][r] *= sc; oc[1][r] *= sc; oc[2][r] *= sc; oc[3][r] *= sc;
            int row = r + (half ? 8 : 0);
            pw[row * 32 + lm]      = f2bf(p0);
            pw[row * 32 + 16 + lm] = f2bf(p1);
        }

        // ---- reload P as A-fragment (in-wave LDS ordering suffices) ----
        Frag pa;
#pragma unroll
        for (int v = 0; v < 8; ++v)
            pa.u[v] = *(const unsigned*)(pw + lm * 32 + kbaseA(v, half));

        // ---- ctx += P x V (four 16-wide hd chunks) ----
#pragma unroll
        for (int f = 0; f < 4; ++f) {
            Frag vb;
            const int vrow = (f * 16 + lm) * 32;     // B lane = hd column
#pragma unroll
            for (int v = 0; v < 8; ++v)
                vb.u[v] = *(const unsigned*)(vc + vrow + 2 * v + half * 16);
            oc[f] = wmma_bf16(pa, vb, oc[f]);
        }
        __syncthreads();    // all reads done before this buffer is re-filled
    }

    // ---- normalize, write ctx [B*S, D] bf16 ----
#pragma unroll
    for (int f = 0; f < 4; ++f)
#pragma unroll
        for (int r = 0; r < 8; ++r) {
            int row = r + (half ? 8 : 0);
            long rg = (long)b * SS + qt * 16 + row;
            int col = h * HD + f * 16 + lm;
            C[rg * DD + col] = f2bf(oc[f][r] / lrun[r]);
        }
}

// ---------------------------------------------------------------------------
// Kernel 3: output projection (bf16 ctx x Wo -> fp32), double-buffered TDM.
// grid = (B*S/64, D/64), block = 128.
// ---------------------------------------------------------------------------
__global__ void __launch_bounds__(128)
mha_out_proj(const unsigned short* __restrict__ ctx, const unsigned* __restrict__ Wop,
             float* __restrict__ out) {
    const int tid  = threadIdx.x;
    const int wid  = tid >> 5;
    const int lane = tid & 31;
    const int half = lane >> 4;
    const int lm   = lane & 15;

    const int m0 = blockIdx.x * 64 + wid * 16;
    const int nb = blockIdx.y * 64;

    __shared__ unsigned wt[2][16 * 64];

    v8f acc[4];
#pragma unroll
    for (int f = 0; f < 4; ++f) acc[f] = (v8f){0,0,0,0,0,0,0,0};

#if HAVE_TDM
    tdm_load_2d(lds_off(&wt[0][0]) + wid * 4 * 64 * 4,
                Wop + (wid * 4) * DD + nb, 2, DD, DD / 2, DD, 64, 4);
#endif

#pragma unroll 1
    for (int k0 = 0; k0 < DD; k0 += 32) {
        const int cur = (k0 >> 5) & 1;
        const unsigned* wcur = wt[cur];
#if HAVE_TDM
        if (k0 + 32 < DD) {
            tdm_load_2d(lds_off(&wt[cur ^ 1][0]) + wid * 4 * 64 * 4,
                        Wop + (((k0 + 32) >> 1) + wid * 4) * DD + nb,
                        2, DD, DD / 2, DD, 64, 4);
            __builtin_amdgcn_s_wait_tensorcnt(1);
        } else {
            __builtin_amdgcn_s_wait_tensorcnt(0);
        }
#else
#pragma unroll
        for (int i = 0; i < 8; ++i) {
            int idx = i * 128 + tid;
            wt[cur][idx] = Wop[((k0 >> 1) + (idx >> 6)) * DD + nb + (idx & 63)];
        }
#endif
        __syncthreads();

        if (k0 + 32 < DD)
            __builtin_prefetch(ctx + (long)(m0 + lm) * DD + k0 + 32, 0, 3);

        Frag a;
#pragma unroll
        for (int v = 0; v < 8; ++v)
            a.u[v] = *(const unsigned*)(ctx + (long)(m0 + lm) * DD + k0 + kbaseA(v, half));
#pragma unroll
        for (int f = 0; f < 4; ++f) {
            Frag b;
#pragma unroll
            for (int v = 0; v < 8; ++v)
                b.u[v] = wcur[(v + half * 8) * 64 + f * 16 + lm];
            acc[f] = wmma_bf16(a, b, acc[f]);
        }
        __syncthreads();
    }

#pragma unroll
    for (int f = 0; f < 4; ++f) {
        const int n = nb + f * 16 + lm;
#pragma unroll
        for (int r = 0; r < 8; ++r) {
            int row = r + (half ? 8 : 0);
            long rg = m0 + row;
            out[rg * DD + n] = acc[f][r];
        }
    }
}

// ---------------------------------------------------------------------------
// launcher
// ---------------------------------------------------------------------------
extern "C" void kernel_launch(void* const* d_in, const int* in_sizes, int n_in,
                              void* d_out, int out_size, void* d_ws, size_t ws_size,
                              hipStream_t stream) {
    const float* q  = (const float*)d_in[0];
    const float* k  = (const float*)d_in[1];
    const float* v  = (const float*)d_in[2];
    const float* Wq = (const float*)d_in[3];
    const float* Wk = (const float*)d_in[4];
    const float* Wv = (const float*)d_in[5];
    const float* Wo = (const float*)d_in[6];

    const size_t actElems = (size_t)BB * HH * SS * HD;   // 4,194,304 bf16
    const size_t wElems   = (size_t)(DD / 2) * DD;       // 524,288 u32

    char* ws = (char*)d_ws;
    unsigned short* qws  = (unsigned short*)ws;                 ws += actElems * 2;
    unsigned short* kws  = (unsigned short*)ws;                 ws += actElems * 2;
    unsigned short* vtws = (unsigned short*)ws;                 ws += actElems * 2;
    unsigned short* cws  = (unsigned short*)ws;                 ws += actElems * 2;
    unsigned* wq = (unsigned*)ws;                               ws += wElems * 4;
    unsigned* wk = (unsigned*)ws;                               ws += wElems * 4;
    unsigned* wv = (unsigned*)ws;                               ws += wElems * 4;
    unsigned* wo = (unsigned*)ws;

    const int cvtBlocks = (int)(wElems / 256);
    mha_cvt_wpair<<<cvtBlocks, 256, 0, stream>>>(Wq, wq);
    mha_cvt_wpair<<<cvtBlocks, 256, 0, stream>>>(Wk, wk);
    mha_cvt_wpair<<<cvtBlocks, 256, 0, stream>>>(Wv, wv);
    mha_cvt_wpair<<<cvtBlocks, 256, 0, stream>>>(Wo, wo);

    mha_proj_qkv<<<dim3((BB * SS) / 64, DD / 64, 3), 128, 0, stream>>>(
        q, k, v, wq, wk, wv, qws, kws, vtws);

    mha_attn<<<dim3((BB * HH * (SS / 16)) / 4), 128, 0, stream>>>(
        qws, kws, vtws, cws);

    mha_out_proj<<<dim3((BB * SS) / 64, DD / 64), 128, 0, stream>>>(
        cws, wo, (float*)d_out);
}
